// GATModelBasic_52561809768869
// MI455X (gfx1250) — matrix operand order repaired and checked by reading, not compiled, not run on hardware
//
#include <hip/hip_runtime.h>
#include <hip/hip_bf16.h>
#include <math.h>

// ---------------------------------------------------------------------------
// GAT 2-layer model for MI455X (gfx1250, wave32).
// GEMMs use v_wmma_f32_16x16x32_bf16 (f32 accumulate). Edge phases use
// float atomics (max via int-ordering trick, add via global_atomic_add_f32).
// ---------------------------------------------------------------------------

#define GN 50000
#define GE 800000
#define GH 8
#define GFIN 128
#define GHID 32
#define GOUT 40
#define ETOT (GE + GN)      // edges + self loops = 850000
#define NEG_SLOPE 0.2f

typedef __attribute__((ext_vector_type(16))) __bf16 v16bf;
typedef __attribute__((ext_vector_type(8)))  __bf16 v8bf;
typedef __attribute__((ext_vector_type(8)))  float  v8f;

static inline int cdiv(long long a, long long b) { return (int)((a + b - 1) / b); }

// ---------------- helpers ----------------

__device__ __forceinline__ float leaky(float v) {
    return v > 0.0f ? v : NEG_SLOPE * v;
}

__device__ __forceinline__ void atomicMaxF(float* addr, float v) {
    // IEEE ordering trick: int max for >=0, unsigned min for <0.
    if (v >= 0.0f) atomicMax((int*)addr, __float_as_int(v));
    else           atomicMin((unsigned int*)addr, __float_as_uint(v));
}

// ---------------- conversion / fill kernels ----------------

__global__ void k_cvt_bf16(const float* __restrict__ in, __bf16* __restrict__ out, int n) {
    int i = blockIdx.x * blockDim.x + threadIdx.x;
    if (i < n) out[i] = (__bf16)in[i];
}

// WT[n*K + k] = (bf16) W[k*Nc + n]
__global__ void k_transpose_bf16(const float* __restrict__ W, __bf16* __restrict__ WT,
                                 int K, int Nc) {
    int i = blockIdx.x * blockDim.x + threadIdx.x;
    if (i >= K * Nc) return;
    int k = i / Nc, n = i % Nc;
    WT[n * K + k] = (__bf16)W[i];
}

__global__ void k_fill(float* __restrict__ p, float v, int n) {
    int i = blockIdx.x * blockDim.x + threadIdx.x;
    if (i < n) p[i] = v;
}

// ---------------- WMMA GEMM, layer 1: [N,128] x [128,256] -> [N,256] ----------------
// Block = 128 threads = 4 waves. Block owns 16 rows; wave w owns cols [w*64, w*64+64).
// M = 50000 = 3125*16 exactly -> no masking; EXEC stays all-ones for WMMA.

__global__ __launch_bounds__(128)
void k_gemm1(const __bf16* __restrict__ A, const __bf16* __restrict__ BT,
             float* __restrict__ C) {
    const int mbase = blockIdx.x * 16;
    const int wave  = threadIdx.x >> 5;
    const int lane  = threadIdx.x & 31;
    const int lhi   = (lane & 16) ? 1 : 0;
    const int l15   = lane & 15;
    const int nbase = wave * 64;
    const int mrow  = mbase + l15;

    v8f acc[4] = {};

    for (int kt = 0; kt < 4; ++kt) {
        // A fragment: row mrow; lanes0-15 K = kt*32+{0..15}, lanes16-31 +16
        const int kbA = kt * 32 + (lhi ? 16 : 0);
        const __bf16* arow = A + mrow * GFIN + kbA;
        v16bf afrag;
        ((v8bf*)&afrag)[0] = *(const v8bf*)(arow);
        ((v8bf*)&afrag)[1] = *(const v8bf*)(arow + 8);

        const int kbB = kt * 32 + (lhi ? 16 : 0);
#pragma unroll
        for (int t = 0; t < 4; ++t) {
            const int ncol = nbase + t * 16 + l15;
            // B fragment: col ncol; 16 contiguous K from transposed weight
            v16bf bfrag = *(const v16bf*)(BT + ncol * GFIN + kbB);
            acc[t] = __builtin_amdgcn_wmma_f32_16x16x32_bf16(
                false, afrag, false, bfrag, (short)0, acc[t], false, false);
        }
    }

    const int mhi = mbase + (lhi ? 8 : 0);
#pragma unroll
    for (int t = 0; t < 4; ++t) {
        const int ncol = nbase + t * 16 + l15;
#pragma unroll
        for (int r = 0; r < 8; ++r)
            C[(mhi + r) * 256 + ncol] = acc[t][r];
    }
}

// ---------------- WMMA GEMM, layer 2: [N,32] x [32,320] -> [N,320] ----------------
// Block = 128 threads = 4 waves; wave w owns cols [w*80, w*80+80) (5 tiles). K=32, one WMMA depth.

__global__ __launch_bounds__(128)
void k_gemm2(const __bf16* __restrict__ A, const __bf16* __restrict__ BT,
             float* __restrict__ C) {
    const int mbase = blockIdx.x * 16;
    const int wave  = threadIdx.x >> 5;
    const int lane  = threadIdx.x & 31;
    const int lhi   = (lane & 16) ? 1 : 0;
    const int l15   = lane & 15;
    const int nbase = wave * 80;
    const int mrow  = mbase + l15;

    const int kbA = lhi ? 16 : 0;
    const __bf16* arow = A + mrow * GHID + kbA;
    v16bf afrag;
    ((v8bf*)&afrag)[0] = *(const v8bf*)(arow);
    ((v8bf*)&afrag)[1] = *(const v8bf*)(arow + 8);

    const int kbB = lhi ? 16 : 0;
    v8f acc[5] = {};
#pragma unroll
    for (int t = 0; t < 5; ++t) {
        const int ncol = nbase + t * 16 + l15;
        v16bf bfrag = *(const v16bf*)(BT + ncol * GHID + kbB);
        acc[t] = __builtin_amdgcn_wmma_f32_16x16x32_bf16(
            false, afrag, false, bfrag, (short)0, acc[t], false, false);
    }

    const int mhi = mbase + (lhi ? 8 : 0);
#pragma unroll
    for (int t = 0; t < 5; ++t) {
        const int ncol = nbase + t * 16 + l15;
#pragma unroll
        for (int r = 0; r < 8; ++r)
            C[(mhi + r) * 320 + ncol] = acc[t][r];
    }
}

// ---------------- attention scores: a_s[n,h] = <h[n,h,:], att_src[h,:]> ----------------

__global__ void k_scores(const float* __restrict__ h, const float* __restrict__ att_src,
                         const float* __restrict__ att_dst, float* __restrict__ a_s,
                         float* __restrict__ a_d, int C) {
    int i = blockIdx.x * blockDim.x + threadIdx.x;
    if (i >= GN * GH) return;
    int head = i & 7;
    const float* hp  = h + (long long)i * C;
    const float* asp = att_src + head * C;
    const float* adp = att_dst + head * C;
    float s = 0.0f, d = 0.0f;
    for (int c = 0; c < C; ++c) {
        float v = hp[c];
        s = fmaf(v, asp[c], s);
        d = fmaf(v, adp[c], d);
    }
    a_s[i] = s;
    a_d[i] = d;
}

// ---------------- edge passes (thread per (edge, head)) ----------------

__device__ __forceinline__ void edge_sd(const int* __restrict__ ei, int e, int& src, int& dst) {
    if (e < GE) { src = ei[e]; dst = ei[GE + e]; }
    else        { src = dst = e - GE; }   // self loop
}

__global__ void k_edge_max(const int* __restrict__ ei, const float* __restrict__ a_s,
                           const float* __restrict__ a_d, float* __restrict__ m) {
    int i = blockIdx.x * blockDim.x + threadIdx.x;
    if (i >= ETOT * GH) return;
    int head = i & 7, e = i >> 3;
    int src, dst; edge_sd(ei, e, src, dst);
    float v = leaky(a_s[src * GH + head] + a_d[dst * GH + head]);
    atomicMaxF(&m[dst * GH + head], v);
}

__global__ void k_edge_expsum(const int* __restrict__ ei, const float* __restrict__ a_s,
                              const float* __restrict__ a_d, const float* __restrict__ m,
                              float* __restrict__ s) {
    int i = blockIdx.x * blockDim.x + threadIdx.x;
    if (i >= ETOT * GH) return;
    int head = i & 7, e = i >> 3;
    int src, dst; edge_sd(ei, e, src, dst);
    float v = leaky(a_s[src * GH + head] + a_d[dst * GH + head]);
    float ex = __expf(v - m[dst * GH + head]);
    atomicAdd(&s[dst * GH + head], ex);
}

__global__ void k_edge_alpha(const int* __restrict__ ei, const float* __restrict__ a_s,
                             const float* __restrict__ a_d, const float* __restrict__ m,
                             const float* __restrict__ s, float* __restrict__ alpha) {
    int i = blockIdx.x * blockDim.x + threadIdx.x;
    if (i >= ETOT * GH) return;
    int head = i & 7, e = i >> 3;
    int src, dst; edge_sd(ei, e, src, dst);
    float v  = leaky(a_s[src * GH + head] + a_d[dst * GH + head]);
    float ex = __expf(v - m[dst * GH + head]);
    alpha[i] = ex / s[dst * GH + head];
}

// thread per (edge, head, channel); out[dst,h,c] += h[src,h,c] * alpha[e,h]
__global__ void k_edge_aggr(const int* __restrict__ ei, const float* __restrict__ alpha,
                            const float* __restrict__ h, float* __restrict__ out, int C) {
    int i = blockIdx.x * blockDim.x + threadIdx.x;
    int total = ETOT * GH * C;
    if (i >= total) return;
    int c = i % C;
    int t = i / C;              // (e*8 + head)
    int head = t & 7, e = t >> 3;
    int src, dst; edge_sd(ei, e, src, dst);
    float a = alpha[t];
    float v = h[(src * GH + head) * C + c] * a;
    atomicAdd(&out[(dst * GH + head) * C + c], v);
}

// ---------------- epilogues ----------------

// mean over heads + bias + relu; produce f32 and bf16 copies for next GEMM
__global__ void k_finalize1(const float* __restrict__ agg, const float* __restrict__ b,
                            float* __restrict__ h2, __bf16* __restrict__ h2b) {
    int i = blockIdx.x * blockDim.x + threadIdx.x;
    if (i >= GN * GHID) return;
    int c = i & 31, node = i >> 5;
    const float* p = agg + node * (GH * GHID);
    float s = 0.0f;
#pragma unroll
    for (int hh = 0; hh < GH; ++hh) s += p[hh * GHID + c];
    s = fmaf(s, 0.125f, b[c]);
    s = fmaxf(s, 0.0f);
    h2[i]  = s;
    h2b[i] = (__bf16)s;
}

// mean over heads + bias, then log_softmax over the 40 outputs; one thread per node
__global__ void k_finalize2(const float* __restrict__ agg, const float* __restrict__ b,
                            float* __restrict__ out) {
    int i = blockIdx.x * blockDim.x + threadIdx.x;
    if (i >= GN) return;
    const float* p = agg + (long long)i * (GH * GOUT);
    float v[GOUT];
    float mx = -INFINITY;
    for (int c = 0; c < GOUT; ++c) {
        float s = 0.0f;
#pragma unroll
        for (int hh = 0; hh < GH; ++hh) s += p[hh * GOUT + c];
        s = fmaf(s, 0.125f, b[c]);
        v[c] = s;
        mx = fmaxf(mx, s);
    }
    float lse = 0.0f;
    for (int c = 0; c < GOUT; ++c) lse += __expf(v[c] - mx);
    lse = mx + __logf(lse);
    for (int c = 0; c < GOUT; ++c) out[i * GOUT + c] = v[c] - lse;
}

// ---------------------------------------------------------------------------

extern "C" void kernel_launch(void* const* d_in, const int* in_sizes, int n_in,
                              void* d_out, int out_size, void* d_ws, size_t ws_size,
                              hipStream_t stream) {
    const float* x        = (const float*)d_in[0];
    const int*   ei       = (const int*)  d_in[1];
    const float* W1       = (const float*)d_in[2];
    const float* att_src1 = (const float*)d_in[3];
    const float* att_dst1 = (const float*)d_in[4];
    const float* b1       = (const float*)d_in[5];
    const float* W2       = (const float*)d_in[6];
    const float* att_src2 = (const float*)d_in[7];
    const float* att_dst2 = (const float*)d_in[8];
    const float* b2       = (const float*)d_in[9];
    float* out = (float*)d_out;

    // ---- workspace layout (256B-aligned slots; big buffers reused per layer) ----
    char* ws = (char*)d_ws;
    size_t off = 0;
    auto carve = [&](size_t bytes) -> char* {
        char* p = ws + off;
        off += (bytes + 255) & ~(size_t)255;
        return p;
    };
    float*  bigA  = (float*) carve((size_t)GN * 320 * 4); // h1 (N,256) / h2full (N,320)
    float*  bigB  = (float*) carve((size_t)GN * 320 * 4); // agg1 / agg2
    __bf16* xb    = (__bf16*)carve((size_t)GN * GFIN * 2);
    __bf16* w1t   = (__bf16*)carve((size_t)256 * GFIN * 2);
    __bf16* w2t   = (__bf16*)carve((size_t)320 * GHID * 2);
    float*  asb   = (float*) carve((size_t)GN * GH * 4);
    float*  adb   = (float*) carve((size_t)GN * GH * 4);
    float*  mb    = (float*) carve((size_t)GN * GH * 4);
    float*  sb    = (float*) carve((size_t)GN * GH * 4);
    float*  alpha = (float*) carve((size_t)ETOT * GH * 4);
    float*  h2    = (float*) carve((size_t)GN * GHID * 4);
    __bf16* h2b   = (__bf16*)carve((size_t)GN * GHID * 2);
    (void)ws_size; (void)n_in; (void)in_sizes; (void)out_size;

    const int TB = 256;
    const int nEH = ETOT * GH;

    // ======================= layer 1 =======================
    k_cvt_bf16<<<cdiv((long long)GN * GFIN, TB), TB, 0, stream>>>(x, xb, GN * GFIN);
    k_transpose_bf16<<<cdiv(GFIN * 256, TB), TB, 0, stream>>>(W1, w1t, GFIN, 256);
    k_gemm1<<<GN / 16, 128, 0, stream>>>(xb, w1t, bigA);

    k_scores<<<cdiv(GN * GH, TB), TB, 0, stream>>>(bigA, att_src1, att_dst1, asb, adb, GHID);

    k_fill<<<cdiv(GN * GH, TB), TB, 0, stream>>>(mb, -INFINITY, GN * GH);
    k_edge_max<<<cdiv(nEH, TB), TB, 0, stream>>>(ei, asb, adb, mb);
    hipMemsetAsync(sb, 0, (size_t)GN * GH * 4, stream);
    k_edge_expsum<<<cdiv(nEH, TB), TB, 0, stream>>>(ei, asb, adb, mb, sb);
    k_edge_alpha<<<cdiv(nEH, TB), TB, 0, stream>>>(ei, asb, adb, mb, sb, alpha);

    hipMemsetAsync(bigB, 0, (size_t)GN * 256 * 4, stream);
    k_edge_aggr<<<cdiv((long long)nEH * GHID, TB), TB, 0, stream>>>(ei, alpha, bigA, bigB, GHID);
    k_finalize1<<<cdiv(GN * GHID, TB), TB, 0, stream>>>(bigB, b1, h2, h2b);

    // ======================= layer 2 =======================
    k_transpose_bf16<<<cdiv(GHID * 320, TB), TB, 0, stream>>>(W2, w2t, GHID, 320);
    k_gemm2<<<GN / 16, 128, 0, stream>>>(h2b, w2t, bigA);

    k_scores<<<cdiv(GN * GH, TB), TB, 0, stream>>>(bigA, att_src2, att_dst2, asb, adb, GOUT);

    k_fill<<<cdiv(GN * GH, TB), TB, 0, stream>>>(mb, -INFINITY, GN * GH);
    k_edge_max<<<cdiv(nEH, TB), TB, 0, stream>>>(ei, asb, adb, mb);
    hipMemsetAsync(sb, 0, (size_t)GN * GH * 4, stream);
    k_edge_expsum<<<cdiv(nEH, TB), TB, 0, stream>>>(ei, asb, adb, mb, sb);
    k_edge_alpha<<<cdiv(nEH, TB), TB, 0, stream>>>(ei, asb, adb, mb, sb, alpha);

    hipMemsetAsync(bigB, 0, (size_t)GN * 320 * 4, stream);
    k_edge_aggr<<<cdiv((long long)nEH * GOUT, TB), TB, 0, stream>>>(ei, alpha, bigA, bigB, GOUT);
    k_finalize2<<<cdiv(GN, TB), TB, 0, stream>>>(bigB, b2, out);
}
